// DeepseekV2DecoderLayer_65816078844552
// MI455X (gfx1250) — matrix operand order repaired and checked by reading, not compiled, not run on hardware
//
#include <hip/hip_runtime.h>
#include <hip/hip_bf16.h>

typedef __bf16 bf16_t;
typedef bf16_t v16bf __attribute__((ext_vector_type(16)));
typedef float  v8f   __attribute__((ext_vector_type(8)));
typedef unsigned int u32;
typedef unsigned short u16;
typedef unsigned int v4u __attribute__((ext_vector_type(4)));
typedef int v4i __attribute__((ext_vector_type(4)));
typedef int v8i __attribute__((ext_vector_type(8)));

#define BNH 32          // B*NH
#define SEQ 2048
#define NTOK 4096       // B*S
#define HD 2048
#define QLORA 1536
#define KVLORA 512
#define ROPE_D 64
#define QHD_D 192
#define VHD_D 128
#define FF_D 8192
#define NHD 16

__device__ __forceinline__ bf16_t f2bf(float f) {
    u32 u = __builtin_bit_cast(u32, f);
    u32 r = u + 0x7fffu + ((u >> 16) & 1u);
    u16 h = (u16)(r >> 16);
    return __builtin_bit_cast(bf16_t, h);
}

__device__ __forceinline__ v8f zero8() {
    v8f z = {0.f,0.f,0.f,0.f,0.f,0.f,0.f,0.f};
    return z;
}

// XOR butterfly within 16-lane groups via v_permlane16_b32 (pure VALU, no LDS)
template<u32 S0, u32 S1>
__device__ __forceinline__ float permx(float v) {
    int s = __builtin_bit_cast(int, v);
    int r = __builtin_amdgcn_permlane16(s, s, (int)S0, (int)S1, false, false);
    return __builtin_bit_cast(float, r);
}
__device__ __forceinline__ float rowmax16(float v) {
    v = fmaxf(v, permx<0x67452301u, 0xEFCDAB89u>(v));   // xor 1
    v = fmaxf(v, permx<0x54761032u, 0xDCFE98BAu>(v));   // xor 2
    v = fmaxf(v, permx<0x32107654u, 0xBA98FEDCu>(v));   // xor 4
    v = fmaxf(v, permx<0xFEDCBA98u, 0x76543210u>(v));   // xor 8
    return v;
}
__device__ __forceinline__ float rowsum16(float v) {
    v += permx<0x67452301u, 0xEFCDAB89u>(v);
    v += permx<0x54761032u, 0xDCFE98BAu>(v);
    v += permx<0x32107654u, 0xBA98FEDCu>(v);
    v += permx<0xFEDCBA98u, 0x76543210u>(v);
    return v;
}

// ---------------------------------------------------------------------------
// TDM: async 2D tile load (128 rows x 32 bf16) global -> LDS (row-major)
// D# group0/group1 packed per CDNA5 ISA 8.3/8.4; groups 2/3 zero (2D tensor).
// This toolchain exposes the 6-arg builtin:
//   (uint32x4 g0, int32x8 g1, int32x4, int32x4, int32x8, i32 cpol)
// ---------------------------------------------------------------------------
__device__ __forceinline__ void tdm_load_tile(u32 lds_off, const bf16_t* gptr,
                                              u32 k_len, u32 rows, u32 stride_elems) {
    unsigned long long ga = (unsigned long long)(size_t)(const void*)gptr;
    v4u g0;
    g0.x = 1u;                                        // count=1, user mode
    g0.y = lds_off;                                   // lds_addr (bytes)
    g0.z = (u32)ga;                                   // global_addr[31:0]
    g0.w = (u32)((ga >> 32) & 0x01ffffffull) | 0x80000000u; // addr[56:32] | type=2
    v8i g1;
    g1[0] = (int)0x00010000u;                         // data_size = 1 (2 bytes)
    g1[1] = (int)((k_len & 0xffffu) << 16);           // tensor_dim0[15:0] @ 63:48
    g1[2] = (int)((k_len >> 16) | ((rows & 0xffffu) << 16)); // dim0 hi | dim1 lo
    g1[3] = (int)((rows >> 16) | (32u << 16));        // dim1 hi | tile_dim0=32
    g1[4] = 128;                                      // tile_dim1=128, tile_dim2=0
    g1[5] = (int)stride_elems;                        // tensor_dim0_stride[31:0]
    g1[6] = 0;                                        // stride hi | dim1_stride lo
    g1[7] = 0;
    v4i z4 = {0, 0, 0, 0};
    v8i z8 = {0, 0, 0, 0, 0, 0, 0, 0};
    __builtin_amdgcn_tensor_load_to_lds(g0, g1, z4, z4, z8, 0);
}

// ---------------------------------------------------------------------------
// Weight convert + transpose: W[K,N] f32 -> WT[NPAD,K] bf16 (zero padded rows)
// ---------------------------------------------------------------------------
__global__ __launch_bounds__(256)
void convert_wT_kernel(const float* __restrict__ W, bf16_t* __restrict__ WT,
                       int K, int N, int NPAD) {
    size_t idx = (size_t)blockIdx.x * 256 + threadIdx.x;
    size_t total = (size_t)NPAD * K;
    if (idx >= total) return;
    int n = (int)(idx / K);
    int k = (int)(idx % K);
    WT[idx] = (n < N) ? f2bf(W[(size_t)k * N + n]) : f2bf(0.f);
}

// ---------------------------------------------------------------------------
// RMSNorm: in f32 [rows, ld_in] (first D cols) -> out bf16 [rows, ld_out]
// ---------------------------------------------------------------------------
__global__ __launch_bounds__(256)
void rmsnorm_bf16_kernel(const float* __restrict__ in, int ld_in, int D,
                         const float* __restrict__ w,
                         bf16_t* __restrict__ out, int ld_out) {
    int row = blockIdx.x;
    const float* x = in + (size_t)row * ld_in;
    float ss = 0.f;
    for (int i = threadIdx.x; i < D; i += 256) { float v = x[i]; ss += v * v; }
    __shared__ float red[256];
    red[threadIdx.x] = ss;
    __syncthreads();
    for (int s = 128; s > 0; s >>= 1) {
        if (threadIdx.x < s) red[threadIdx.x] += red[threadIdx.x + s];
        __syncthreads();
    }
    float inv = rsqrtf(red[0] / (float)D + 1e-6f);
    for (int i = threadIdx.x; i < D; i += 256)
        out[(size_t)row * ld_out + i] = f2bf(w[i] * x[i] * inv);
}

// ---------------------------------------------------------------------------
// WMMA bf16 GEMM with TDM tile staging:
//   C[M,N] = A[M,K](bf16 row-major) @ WT[NPAD,K](bf16, pre-transposed)
// BM=128 BN=128 BK=32; 256 thr = 8 waves (4M x 2N), wave tile 32x64.
// EPI: 0 plain f32, 1 f32 + residual, 2 fused gate/up silu -> bf16
// ---------------------------------------------------------------------------
enum { EPI_PLAIN = 0, EPI_ADD = 1, EPI_GATEUP = 2 };

template<int EPI>
__global__ __launch_bounds__(256)
void gemm_bf16_wmma(const bf16_t* __restrict__ A,
                    const bf16_t* __restrict__ WT0,
                    const bf16_t* __restrict__ WT1,
                    const float* __restrict__ R,
                    float* __restrict__ Cf,
                    bf16_t* __restrict__ Cb,
                    int M, int N, int K) {
    __shared__ bf16_t As[128 * 32];
    __shared__ bf16_t Bs0[128 * 32];                       // [n][k]
    __shared__ bf16_t Bs1[(EPI == EPI_GATEUP) ? 128 * 32 : 16];

    const int tid  = threadIdx.x;
    const int lane = tid & 31;
    const int wrow = (tid >> 5) & 3;     // 0..3 (M)
    const int wcol = tid >> 7;           // 0..1 (N)
    const int half = lane >> 4;
    const int l16  = lane & 15;

    const u32 as_off  = (u32)(size_t)(void*)As;
    const u32 bs0_off = (u32)(size_t)(void*)Bs0;
    const u32 bs1_off = (u32)(size_t)(void*)Bs1;

    v8f acc0[2][4];
    v8f acc1[2][4];
    #pragma unroll
    for (int mt = 0; mt < 2; ++mt)
        #pragma unroll
        for (int nt = 0; nt < 4; ++nt) { acc0[mt][nt] = zero8(); acc1[mt][nt] = zero8(); }

    const size_t rowBase = (size_t)blockIdx.y * 128;
    const size_t colBase = (size_t)blockIdx.x * 128;

    for (int k0 = 0; k0 < K; k0 += 32) {
        if (tid < 32) {      // wave 0 drives the Tensor Data Mover
            tdm_load_tile(as_off,  A   + rowBase * K + k0, (u32)K, 128u, (u32)K);
            tdm_load_tile(bs0_off, WT0 + colBase * K + k0, (u32)K, 128u, (u32)K);
            if constexpr (EPI == EPI_GATEUP)
                tdm_load_tile(bs1_off, WT1 + colBase * K + k0, (u32)K, 128u, (u32)K);
            __builtin_amdgcn_s_wait_tensorcnt(0);
        }
        __syncthreads();

        // A fragments (2 m-tiles)
        v16bf af[2];
        #pragma unroll
        for (int mt = 0; mt < 2; ++mt) {
            union { v16bf v; u32 u[8]; } t;
            const bf16_t* base = As + (wrow * 32 + mt * 16 + l16) * 32;
            const u32* p  = (const u32*)(base + half * 16);
            const u32* p2 = (const u32*)(base + half * 16 + 8);
            #pragma unroll
            for (int q = 0; q < 4; ++q) { t.u[q] = p[q]; t.u[4 + q] = p2[q]; }
            af[mt] = t.v;
        }
        // B fragments (4 n-tiles) + WMMA
        #pragma unroll
        for (int nt = 0; nt < 4; ++nt) {
            int nc = wcol * 64 + nt * 16 + l16;
            union { v16bf v; u32 u[8]; } b0;
            const u32* p = (const u32*)(Bs0 + nc * 32 + half * 16);
            #pragma unroll
            for (int q = 0; q < 8; ++q) b0.u[q] = p[q];
            #pragma unroll
            for (int mt = 0; mt < 2; ++mt)
                acc0[mt][nt] = __builtin_amdgcn_wmma_f32_16x16x32_bf16(
                    false, af[mt], false, b0.v, (short)0, acc0[mt][nt], false, false);
            if constexpr (EPI == EPI_GATEUP) {
                union { v16bf v; u32 u[8]; } b1;
                const u32* p1 = (const u32*)(Bs1 + nc * 32 + half * 16);
                #pragma unroll
                for (int q = 0; q < 8; ++q) b1.u[q] = p1[q];
                #pragma unroll
                for (int mt = 0; mt < 2; ++mt)
                    acc1[mt][nt] = __builtin_amdgcn_wmma_f32_16x16x32_bf16(
                        false, af[mt], false, b1.v, (short)0, acc1[mt][nt], false, false);
            }
        }
        __syncthreads();
    }

    // epilogue
    #pragma unroll
    for (int mt = 0; mt < 2; ++mt) {
        #pragma unroll
        for (int nt = 0; nt < 4; ++nt) {
            int rb = (int)rowBase + wrow * 32 + mt * 16 + half * 8;
            int cb = (int)colBase + wcol * 64 + nt * 16 + l16;
            if (cb < N) {
                #pragma unroll
                for (int r = 0; r < 8; ++r) {
                    size_t idx = (size_t)(rb + r) * N + cb;
                    float v = acc0[mt][nt][r];
                    if constexpr (EPI == EPI_GATEUP) {
                        float g = v;
                        float u = acc1[mt][nt][r];
                        float s = g / (1.f + __expf(-g));
                        Cb[idx] = f2bf(s * u);
                    } else if constexpr (EPI == EPI_ADD) {
                        Cf[idx] = v + R[idx];
                    } else {
                        Cf[idx] = v;
                    }
                }
            }
        }
    }
}

// ---------------------------------------------------------------------------
// RoPE + head assembly: qs[bh][s][192], ks[bh][s][192], vT[bh][128][s] (bf16)
// ---------------------------------------------------------------------------
__global__ __launch_bounds__(256)
void rope_assemble_kernel(const float* __restrict__ q,
                          const float* __restrict__ ckv,
                          const float* __restrict__ kv,
                          const float* __restrict__ cosT,
                          const float* __restrict__ sinT,
                          const int* __restrict__ pos,
                          bf16_t* __restrict__ qs,
                          bf16_t* __restrict__ ks,
                          bf16_t* __restrict__ vT) {
    int tok = blockIdx.x;
    int h   = blockIdx.y;
    int b   = tok / SEQ, s = tok % SEQ;
    int p   = pos[tok];
    int t   = threadIdx.x;
    size_t bh = (size_t)(b * NHD + h);
    bf16_t* qrow = qs + (bh * SEQ + s) * QHD_D;
    bf16_t* krow = ks + (bh * SEQ + s) * QHD_D;
    const float* qh  = q  + (size_t)tok * (NHD * QHD_D) + h * QHD_D;
    const float* kvh = kv + (size_t)tok * (NHD * 256)   + h * 256;
    if (t < 128) {
        qrow[t] = f2bf(qh[t]);          // q_nope
        krow[t] = f2bf(kvh[t]);         // k_nope
        vT[(bh * VHD_D + t) * SEQ + s] = f2bf(kvh[128 + t]);
    } else if (t < 192) {
        int j  = t - 128;               // 0..63
        int jj = j & 31;
        float c  = cosT[(size_t)p * ROPE_D + j];
        float sn = sinT[(size_t)p * ROPE_D + j];
        const float* qpe = qh + 128;
        const float* kpe = ckv + (size_t)tok * 576 + 512;
        float qv, kvv;
        if (j < 32) {
            qv  = qpe[2 * jj] * c - qpe[2 * jj + 1] * sn;
            kvv = kpe[2 * jj] * c - kpe[2 * jj + 1] * sn;
        } else {
            qv  = qpe[2 * jj + 1] * c + qpe[2 * jj] * sn;
            kvv = kpe[2 * jj + 1] * c + kpe[2 * jj] * sn;
        }
        qrow[128 + j] = f2bf(qv);
        krow[128 + j] = f2bf(kvv);
    }
}

// ---------------------------------------------------------------------------
// Causal MLA attention (flash-style, WMMA bf16), one wave per 16 query rows.
// ---------------------------------------------------------------------------
__global__ __launch_bounds__(256)
void mla_attention_kernel(const bf16_t* __restrict__ qs,
                          const bf16_t* __restrict__ ks,
                          const bf16_t* __restrict__ vT,
                          bf16_t* __restrict__ attn) {
    __shared__ bf16_t pT[8][16 * 32];
    const int wid  = threadIdx.x >> 5;
    const int lane = threadIdx.x & 31;
    const int half = lane >> 4;
    const int l16  = lane & 15;

    const int tileIdx      = blockIdx.x * 8 + wid;
    const int tilesPerHead = SEQ / 16;
    const int bh = tileIdx / tilesPerHead;
    const int qb = (tileIdx % tilesPerHead) * 16;

    const bf16_t* Q  = qs + (size_t)bh * SEQ * QHD_D;
    const bf16_t* Kp = ks + (size_t)bh * SEQ * QHD_D;
    const bf16_t* Vt = vT + (size_t)bh * VHD_D * SEQ;

    v16bf qf[6];
    {
        const bf16_t* qr = Q + (size_t)(qb + l16) * QHD_D;
        #pragma unroll
        for (int t = 0; t < 6; ++t) {
            union { v16bf v; u32 u[8]; } tmp;
            const u32* p  = (const u32*)(qr + t * 32 + half * 16);
            const u32* p2 = (const u32*)(qr + t * 32 + half * 16 + 8);
            #pragma unroll
            for (int i = 0; i < 4; ++i) { tmp.u[i] = p[i]; tmp.u[4 + i] = p2[i]; }
            qf[t] = tmp.v;
        }
    }

    float m_i[8], l_i[8];
    v8f o[8];
    #pragma unroll
    for (int r = 0; r < 8; ++r) { m_i[r] = -3.0e38f; l_i[r] = 0.f; }
    #pragma unroll
    for (int dt = 0; dt < 8; ++dt) o[dt] = zero8();

    const float scale = 0.07216878364870323f;   // 192^-0.5
    const int kend = qb + 16;

    for (int k0 = 0; k0 < kend; k0 += 32) {
        float pv[2][8];
        #pragma unroll
        for (int kt = 0; kt < 2; ++kt) {
            int kb = k0 + kt * 16;
            if (kb < kend) {
                v8f sacc = zero8();
                const bf16_t* kr = Kp + (size_t)(kb + l16) * QHD_D;
                #pragma unroll
                for (int t = 0; t < 6; ++t) {
                    union { v16bf v; u32 u[8]; } kf;
                    const u32* p = (const u32*)(kr + t * 32 + half * 16);
                    #pragma unroll
                    for (int i = 0; i < 8; ++i) kf.u[i] = p[i];
                    sacc = __builtin_amdgcn_wmma_f32_16x16x32_bf16(
                        false, qf[t], false, kf.v, (short)0, sacc, false, false);
                }
                int kj = kb + l16;
                #pragma unroll
                for (int r = 0; r < 8; ++r) {
                    int qi = qb + half * 8 + r;
                    float v = sacc[r] * scale;
                    pv[kt][r] = (kj > qi) ? -3.0e38f : v;
                }
            } else {
                #pragma unroll
                for (int r = 0; r < 8; ++r) pv[kt][r] = -3.0e38f;
            }
        }
        // online softmax stats (VALU permlane16 butterflies, no LDS)
        float a_r[8];
        #pragma unroll
        for (int r = 0; r < 8; ++r) {
            float v = rowmax16(fmaxf(pv[0][r], pv[1][r]));
            float mn = fmaxf(m_i[r], v);
            float alpha = __expf(m_i[r] - mn);
            float p0 = (pv[0][r] > -1.0e37f) ? __expf(pv[0][r] - mn) : 0.f;
            float p1 = (pv[1][r] > -1.0e37f) ? __expf(pv[1][r] - mn) : 0.f;
            float sum = rowsum16(p0 + p1);
            l_i[r] = l_i[r] * alpha + sum;
            m_i[r] = mn;
            a_r[r] = alpha;
            pv[0][r] = p0; pv[1][r] = p1;
        }
        #pragma unroll
        for (int dt = 0; dt < 8; ++dt)
            #pragma unroll
            for (int r = 0; r < 8; ++r) o[dt][r] *= a_r[r];
        // bounce P through per-wave LDS slab to re-stripe into A-fragment layout
        bf16_t* pt = pT[wid];
        #pragma unroll
        for (int kt = 0; kt < 2; ++kt)
            #pragma unroll
            for (int r = 0; r < 8; ++r)
                pt[(r + 8 * half) * 32 + kt * 16 + l16] = f2bf(pv[kt][r]);
        asm volatile("s_wait_dscnt 0" ::: "memory");
        union { v16bf v; u32 u[8]; } pf;
        {
            const u32* p  = (const u32*)(pt + l16 * 32 + half * 16);
            const u32* p2 = (const u32*)(pt + l16 * 32 + half * 16 + 8);
            #pragma unroll
            for (int i = 0; i < 4; ++i) { pf.u[i] = p[i]; pf.u[4 + i] = p2[i]; }
        }
        asm volatile("s_wait_dscnt 0" ::: "memory");
        // P @ V (V transposed -> contiguous lane loads)
        #pragma unroll
        for (int dt = 0; dt < 8; ++dt) {
            union { v16bf v; u32 u[8]; } vf;
            const bf16_t* vr = Vt + (size_t)(dt * 16 + l16) * SEQ + k0 + half * 16;
            #pragma unroll
            for (int i = 0; i < 8; ++i) vf.u[i] = ((const u32*)vr)[i];
            o[dt] = __builtin_amdgcn_wmma_f32_16x16x32_bf16(
                false, pf.v, false, vf.v, (short)0, o[dt], false, false);
        }
    }

    float invl[8];
    #pragma unroll
    for (int r = 0; r < 8; ++r) invl[r] = 1.f / l_i[r];
    int b = bh >> 4, h = bh & 15;
    #pragma unroll
    for (int dt = 0; dt < 8; ++dt) {
        #pragma unroll
        for (int r = 0; r < 8; ++r) {
            int row = qb + r + 8 * half;
            int col = dt * 16 + l16;
            attn[((size_t)(b * SEQ + row)) * (NHD * VHD_D) + h * VHD_D + col] =
                f2bf(o[dt][r] * invl[r]);
        }
    }
}

// ---------------------------------------------------------------------------
// Orchestration
// ---------------------------------------------------------------------------
extern "C" void kernel_launch(void* const* d_in, const int* in_sizes, int n_in,
                              void* d_out, int out_size, void* d_ws, size_t ws_size,
                              hipStream_t stream) {
    const float* hidden = (const float*)d_in[0];
    const float* ln1_w  = (const float*)d_in[1];
    const float* q_a_k  = (const float*)d_in[2];
    const float* q_a_ln = (const float*)d_in[3];
    const float* q_b_k  = (const float*)d_in[4];
    const float* kv_a_k = (const float*)d_in[5];
    const float* kv_a_ln= (const float*)d_in[6];
    const float* kv_b_k = (const float*)d_in[7];
    const float* o_k    = (const float*)d_in[8];
    const float* ln2_w  = (const float*)d_in[9];
    const float* gate_k = (const float*)d_in[10];
    const float* up_k   = (const float*)d_in[11];
    const float* down_k = (const float*)d_in[12];
    const float* sinT   = (const float*)d_in[13];
    const float* cosT   = (const float*)d_in[14];
    const int*   pos    = (const int*)d_in[15];
    float* out = (float*)d_out;

    char* ws = (char*)d_ws;
    const size_t o_xbf  = 0;
    const size_t o_h1   = o_xbf  + (size_t)NTOK*HD*2;
    const size_t o_x2   = o_h1   + (size_t)NTOK*HD*4;
    const size_t o_attn = o_x2   + (size_t)NTOK*HD*2;
    const size_t o_scr  = o_attn + (size_t)NTOK*HD*2;
    // phase-1 scratch
    const size_t o_qa   = o_scr;
    const size_t o_qan  = o_qa   + (size_t)NTOK*QLORA*4;
    const size_t o_q    = o_qan  + (size_t)NTOK*QLORA*2;
    const size_t o_ckv  = o_q    + (size_t)NTOK*NHD*QHD_D*4;
    const size_t o_ckvn = o_ckv  + (size_t)NTOK*576*4;
    const size_t o_kv   = o_ckvn + (size_t)NTOK*KVLORA*2;
    const size_t o_qs   = o_kv   + (size_t)NTOK*NHD*256*4;
    const size_t o_ks   = o_qs   + (size_t)BNH*SEQ*QHD_D*2;
    const size_t o_vt   = o_ks   + (size_t)BNH*SEQ*QHD_D*2;
    const size_t o_mlp  = o_scr;            // phase-2 overlays phase-1
    // bf16 transposed weights (persist across phases)
    const size_t o_w    = o_vt   + (size_t)BNH*VHD_D*SEQ*2;
    const size_t o_qaT  = o_w;
    const size_t o_qbT  = o_qaT + (size_t)QLORA*HD*2;
    const size_t o_kvaT = o_qbT + (size_t)(NHD*QHD_D)*QLORA*2;
    const size_t o_kvbT = o_kvaT+ (size_t)640*HD*2;             // 576 -> 640 pad
    const size_t o_oT   = o_kvbT+ (size_t)(NHD*256)*KVLORA*2;
    const size_t o_gT   = o_oT  + (size_t)HD*HD*2;
    const size_t o_uT   = o_gT  + (size_t)FF_D*HD*2;
    const size_t o_dT   = o_uT  + (size_t)FF_D*HD*2;

    bf16_t* x_bf  = (bf16_t*)(ws + o_xbf);
    float*  h1    = (float*) (ws + o_h1);
    bf16_t* x2    = (bf16_t*)(ws + o_x2);
    bf16_t* attn  = (bf16_t*)(ws + o_attn);
    float*  q_a   = (float*) (ws + o_qa);
    bf16_t* q_an  = (bf16_t*)(ws + o_qan);
    float*  q     = (float*) (ws + o_q);
    float*  ckv   = (float*) (ws + o_ckv);
    bf16_t* ckvn  = (bf16_t*)(ws + o_ckvn);
    float*  kv    = (float*) (ws + o_kv);
    bf16_t* qs    = (bf16_t*)(ws + o_qs);
    bf16_t* ks    = (bf16_t*)(ws + o_ks);
    bf16_t* vT    = (bf16_t*)(ws + o_vt);
    bf16_t* mlp   = (bf16_t*)(ws + o_mlp);
    bf16_t* qaT   = (bf16_t*)(ws + o_qaT);
    bf16_t* qbT   = (bf16_t*)(ws + o_qbT);
    bf16_t* kvaT  = (bf16_t*)(ws + o_kvaT);
    bf16_t* kvbT  = (bf16_t*)(ws + o_kvbT);
    bf16_t* oT    = (bf16_t*)(ws + o_oT);
    bf16_t* gT    = (bf16_t*)(ws + o_gT);
    bf16_t* uT    = (bf16_t*)(ws + o_uT);
    bf16_t* dT    = (bf16_t*)(ws + o_dT);

    dim3 blk(256);
    auto cvGrid = [](size_t total) { return dim3((unsigned)((total + 255) / 256)); };

    // 0. one-time weight convert + transpose (f32 [K,N] -> bf16 [NPAD,K])
    convert_wT_kernel<<<cvGrid((size_t)QLORA*HD), blk, 0, stream>>>(q_a_k, qaT, HD, QLORA, QLORA);
    convert_wT_kernel<<<cvGrid((size_t)(NHD*QHD_D)*QLORA), blk, 0, stream>>>(q_b_k, qbT, QLORA, NHD*QHD_D, NHD*QHD_D);
    convert_wT_kernel<<<cvGrid((size_t)640*HD), blk, 0, stream>>>(kv_a_k, kvaT, HD, 576, 640);
    convert_wT_kernel<<<cvGrid((size_t)(NHD*256)*KVLORA), blk, 0, stream>>>(kv_b_k, kvbT, KVLORA, NHD*256, NHD*256);
    convert_wT_kernel<<<cvGrid((size_t)HD*HD), blk, 0, stream>>>(o_k, oT, NHD*VHD_D, HD, HD);
    convert_wT_kernel<<<cvGrid((size_t)FF_D*HD), blk, 0, stream>>>(gate_k, gT, HD, FF_D, FF_D);
    convert_wT_kernel<<<cvGrid((size_t)FF_D*HD), blk, 0, stream>>>(up_k, uT, HD, FF_D, FF_D);
    convert_wT_kernel<<<cvGrid((size_t)HD*FF_D), blk, 0, stream>>>(down_k, dT, FF_D, HD, HD);

    // 1. x = rmsnorm(hidden, ln1_w) -> bf16
    rmsnorm_bf16_kernel<<<NTOK, blk, 0, stream>>>(hidden, HD, HD, ln1_w, x_bf, HD);
    // 2. q_a = x @ q_a_kernel
    gemm_bf16_wmma<EPI_PLAIN><<<dim3(QLORA/128, NTOK/128), blk, 0, stream>>>(
        x_bf, qaT, nullptr, nullptr, q_a, nullptr, NTOK, QLORA, HD);
    // 3. rmsnorm(q_a)
    rmsnorm_bf16_kernel<<<NTOK, blk, 0, stream>>>(q_a, QLORA, QLORA, q_a_ln, q_an, QLORA);
    // 4. q = q_an @ q_b_kernel
    gemm_bf16_wmma<EPI_PLAIN><<<dim3(NHD*QHD_D/128, NTOK/128), blk, 0, stream>>>(
        q_an, qbT, nullptr, nullptr, q, nullptr, NTOK, NHD*QHD_D, QLORA);
    // 5. ckv = x @ kv_a_kernel
    gemm_bf16_wmma<EPI_PLAIN><<<dim3(640/128, NTOK/128), blk, 0, stream>>>(
        x_bf, kvaT, nullptr, nullptr, ckv, nullptr, NTOK, 576, HD);
    // 6. rmsnorm(ckv[:, :512])
    rmsnorm_bf16_kernel<<<NTOK, blk, 0, stream>>>(ckv, 576, KVLORA, kv_a_ln, ckvn, KVLORA);
    // 7. kv = ckvn @ kv_b_kernel
    gemm_bf16_wmma<EPI_PLAIN><<<dim3(NHD*256/128, NTOK/128), blk, 0, stream>>>(
        ckvn, kvbT, nullptr, nullptr, kv, nullptr, NTOK, NHD*256, KVLORA);
    // 8. RoPE + assemble qs/ks/vT
    rope_assemble_kernel<<<dim3(NTOK, NHD), blk, 0, stream>>>(
        q, ckv, kv, cosT, sinT, pos, qs, ks, vT);
    // 9. attention
    mla_attention_kernel<<<dim3(BNH * (SEQ/16) / 8), blk, 0, stream>>>(qs, ks, vT, attn);
    // 10. h1 = attn @ o_kernel + hidden
    gemm_bf16_wmma<EPI_ADD><<<dim3(HD/128, NTOK/128), blk, 0, stream>>>(
        attn, oT, nullptr, hidden, h1, nullptr, NTOK, HD, NHD*VHD_D);
    // 11. x2 = rmsnorm(h1, ln2_w)
    rmsnorm_bf16_kernel<<<NTOK, blk, 0, stream>>>(h1, HD, HD, ln2_w, x2, HD);
    // 12. mlp = silu(x2@gate) * (x2@up)  (fused, TDM-staged both weights)
    gemm_bf16_wmma<EPI_GATEUP><<<dim3(FF_D/128, NTOK/128), blk, 0, stream>>>(
        x2, gT, uT, nullptr, nullptr, mlp, NTOK, FF_D, HD);
    // 13. out = mlp @ down + h1
    gemm_bf16_wmma<EPI_ADD><<<dim3(HD/128, NTOK/128), blk, 0, stream>>>(
        mlp, dT, nullptr, h1, out, nullptr, NTOK, HD, FF_D);

    (void)in_sizes; (void)n_in; (void)out_size; (void)ws_size;
}
